// FactorGraphResidual_33535104647628
// MI455X (gfx1250) — matrix operand order repaired and checked by reading, not compiled, not run on hardware
//
#include <hip/hip_runtime.h>

#define N_DIM 8192
#define F_DIM 128

typedef __attribute__((ext_vector_type(16))) __bf16 v16bf;
typedef __attribute__((ext_vector_type(8)))  float  v8f;
typedef __attribute__((ext_vector_type(2)))  float  v2f;

union Frag16 {            // 16 bf16 elements = 8 VGPRs, matches WMMA A/B operand
    v16bf v;
    uint4 q[2];
};

__device__ __forceinline__ unsigned short f2bf(float x) {
    unsigned u = __float_as_uint(x);
    u = u + 0x7FFFu + ((u >> 16) & 1u);          // round-to-nearest-even
    return (unsigned short)(u >> 16);
}
__device__ __forceinline__ float bf2f(unsigned short h) {
    return __uint_as_float(((unsigned)h) << 16);
}

// ---------------------------------------------------------------------------
// Prep: feats [N,F] f32  ->  featsT_hi / featsT_lo [F,N] bf16 (split precision)
// ---------------------------------------------------------------------------
__global__ void prep_featsT(const float* __restrict__ feats,
                            unsigned short* __restrict__ ftHi,
                            unsigned short* __restrict__ ftLo) {
    int idx = blockIdx.x * 256 + threadIdx.x;    // 0 .. N*F-1
    int r = idx >> 7;
    int c = idx & 127;
    float x = feats[idx];
    unsigned short h = f2bf(x);
    unsigned short l = f2bf(x - bf2f(h));
    ftHi[(size_t)c * N_DIM + r] = h;
    ftLo[(size_t)c * N_DIM + r] = l;
}

// ---------------------------------------------------------------------------
// Phase 1: pos_agg = (node_adj>0)@feats, neg_agg = (node_adj<0)@feats,
//          edge_support = edge_adj@feats   -- all via bf16 WMMA, f32 accum.
// One workgroup = 32 output rows x 128 cols, K-loop over 8192 in chunks of 32.
// ---------------------------------------------------------------------------
#define LDS_STRIDE 48   // elements; 96 B rows -> 16B-aligned b128 LDS loads

#define WMMA_BF16(A, B, C) \
    __builtin_amdgcn_wmma_f32_16x16x32_bf16(false, (A), false, (B), (short)0, (C), false, false)

__device__ __forceinline__ Frag16 ldsA(const unsigned short* s, int mbase, int lane) {
    int m  = mbase + (lane & 15);
    int k0 = (lane >> 4) << 4;                   // lanes 0-15: K 0..15, 16-31: K 16..31
    Frag16 f;
    f.q[0] = *(const uint4*)(s + m * LDS_STRIDE + k0);
    f.q[1] = *(const uint4*)(s + m * LDS_STRIDE + k0 + 8);
    return f;
}

__device__ __forceinline__ Frag16 gldB(const unsigned short* ft, int col, int k0) {
    // B 32x16 bf16: lane = column, lanes 0-15 hold K k0..k0+15, lanes 16-31 K k0+16..k0+31
    const uint4* p = (const uint4*)(ft + (size_t)col * N_DIM + k0);
    Frag16 f;
    f.q[0] = p[0];
    f.q[1] = p[1];
    return f;
}

__global__ __launch_bounds__(256) void agg_kernel(
    const float* __restrict__ node_adj, const float* __restrict__ edge_adj,
    const unsigned short* __restrict__ ftHi, const unsigned short* __restrict__ ftLo,
    float* __restrict__ posAgg, float* __restrict__ negAgg, float* __restrict__ edgeAgg) {

    __shared__ __align__(16) unsigned short sPos[32 * LDS_STRIDE];
    __shared__ __align__(16) unsigned short sNeg[32 * LDS_STRIDE];
    __shared__ __align__(16) unsigned short sHi [32 * LDS_STRIDE];
    __shared__ __align__(16) unsigned short sLo [32 * LDS_STRIDE];

    const int tid    = threadIdx.x;
    const int wave   = tid >> 5;
    const int lane   = tid & 31;
    const int m_tile = wave & 1;                 // 0/1 -> rows 0-15 / 16-31
    const int n_base = wave >> 1;                // 0..3 -> col tiles n_base, n_base+4
    const int m0     = blockIdx.x * 32;

    v8f accP[2], accN[2], accE[2];
#pragma unroll
    for (int i = 0; i < 2; ++i) {
        accP[i] = (v8f){0, 0, 0, 0, 0, 0, 0, 0};
        accN[i] = (v8f){0, 0, 0, 0, 0, 0, 0, 0};
        accE[i] = (v8f){0, 0, 0, 0, 0, 0, 0, 0};
    }

    const int r  = tid >> 3;                     // staging row 0..31
    const int c4 = (tid & 7) << 2;               // staging col group {0,4,...,28}

    for (int kk = 0; kk < N_DIM; kk += 32) {
        __syncthreads();                         // previous chunk fully consumed

        const size_t gbase = (size_t)(m0 + r) * N_DIM + kk + c4;
        const float4 na = *(const float4*)(node_adj + gbase);
        const float4 ea = *(const float4*)(edge_adj + gbase);
        if (kk + 32 < N_DIM) {                   // global_prefetch_b8 next K chunk
            __builtin_prefetch(node_adj + gbase + 32, 0, 0);
            __builtin_prefetch(edge_adj + gbase + 32, 0, 0);
        }

        ushort4 p, n, h, l;
        {
            const float nx[4] = {na.x, na.y, na.z, na.w};
            const float ex[4] = {ea.x, ea.y, ea.z, ea.w};
            unsigned short pp[4], nn[4], hh[4], ll[4];
#pragma unroll
            for (int i = 0; i < 4; ++i) {
                pp[i] = (nx[i] > 0.f) ? (unsigned short)0x3F80 : (unsigned short)0;
                nn[i] = (nx[i] < 0.f) ? (unsigned short)0x3F80 : (unsigned short)0;
                hh[i] = f2bf(ex[i]);
                ll[i] = f2bf(ex[i] - bf2f(hh[i]));
            }
            p = make_ushort4(pp[0], pp[1], pp[2], pp[3]);
            n = make_ushort4(nn[0], nn[1], nn[2], nn[3]);
            h = make_ushort4(hh[0], hh[1], hh[2], hh[3]);
            l = make_ushort4(ll[0], ll[1], ll[2], ll[3]);
        }
        const int so = r * LDS_STRIDE + c4;
        *(ushort4*)(sPos + so) = p;
        *(ushort4*)(sNeg + so) = n;
        *(ushort4*)(sHi  + so) = h;
        *(ushort4*)(sLo  + so) = l;

        __syncthreads();

        const Frag16 aP = ldsA(sPos, m_tile * 16, lane);
        const Frag16 aN = ldsA(sNeg, m_tile * 16, lane);
        const Frag16 aH = ldsA(sHi,  m_tile * 16, lane);
        const Frag16 aL = ldsA(sLo,  m_tile * 16, lane);

        const int kcol = kk + ((lane >> 4) << 4);
#pragma unroll
        for (int nt = 0; nt < 2; ++nt) {
            const int col = (n_base + nt * 4) * 16 + (lane & 15);
            const Frag16 bH = gldB(ftHi, col, kcol);
            const Frag16 bL = gldB(ftLo, col, kcol);
            accP[nt] = WMMA_BF16(aP.v, bH.v, accP[nt]);
            accP[nt] = WMMA_BF16(aP.v, bL.v, accP[nt]);
            accN[nt] = WMMA_BF16(aN.v, bH.v, accN[nt]);
            accN[nt] = WMMA_BF16(aN.v, bL.v, accN[nt]);
            accE[nt] = WMMA_BF16(aH.v, bH.v, accE[nt]);
            accE[nt] = WMMA_BF16(aH.v, bL.v, accE[nt]);
            accE[nt] = WMMA_BF16(aL.v, bH.v, accE[nt]);
        }
    }

    // C/D layout: VGPR v -> row v (lanes 0-15) / row v+8 (lanes 16-31), col = lane&15
    const int rbase = m0 + m_tile * 16 + ((lane >> 4) << 3);
#pragma unroll
    for (int nt = 0; nt < 2; ++nt) {
        const int cg = (n_base + nt * 4) * 16 + (lane & 15);
#pragma unroll
        for (int v = 0; v < 8; ++v) {
            const size_t o = (size_t)(rbase + v) * F_DIM + cg;
            posAgg[o]  = accP[nt][v];
            negAgg[o]  = accN[nt][v];
            edgeAgg[o] = accE[nt][v];
        }
    }
}

// ---------------------------------------------------------------------------
// Phase 2: out = [pos|neg]@node_weight + edge_support@edge_weight
//              + node_bias*diag(edge_adj) + edge_bias*diag(node_adj) + feats
// f32 WMMA (16x16x4) -- lossless small GEMMs.
// ---------------------------------------------------------------------------
__global__ __launch_bounds__(256) void out_kernel(
    const float* __restrict__ posAgg, const float* __restrict__ negAgg,
    const float* __restrict__ edgeAgg,
    const float* __restrict__ nodeW, const float* __restrict__ edgeW,
    const float* __restrict__ nodeB, const float* __restrict__ edgeB,
    const float* __restrict__ node_adj, const float* __restrict__ edge_adj,
    const float* __restrict__ feats, float* __restrict__ out) {

    const int tid  = threadIdx.x;
    const int wave = tid >> 5;                   // col tile 0..7
    const int lane = tid & 31;
    const int r0   = blockIdx.x * 16;            // 512 blocks cover 8192 rows
    const int m    = lane & 15;
    const int hi   = lane >> 4;
    const int cg   = wave * 16 + m;

    v8f acc = (v8f){0, 0, 0, 0, 0, 0, 0, 0};

#if __has_builtin(__builtin_amdgcn_wmma_f32_16x16x4_f32)
    // A 16x4 f32: lane m<16 holds K={0,1}; lane m+16 holds K={2,3}
#pragma unroll 4
    for (int k = 0; k < 256; k += 4) {
        const float* Ap = (k < 128)
            ? posAgg + ((size_t)(r0 + m) * F_DIM + k + hi * 2)
            : negAgg + ((size_t)(r0 + m) * F_DIM + (k - 128) + hi * 2);
        v2f a = *(const v2f*)Ap;
        v2f b;
        b.x = nodeW[(size_t)(k + hi * 2)     * F_DIM + cg];
        b.y = nodeW[(size_t)(k + hi * 2 + 1) * F_DIM + cg];
        acc = __builtin_amdgcn_wmma_f32_16x16x4_f32(false, a, false, b, (short)0, acc, false, false);
    }
#pragma unroll 4
    for (int k = 0; k < 128; k += 4) {
        v2f a = *(const v2f*)(edgeAgg + ((size_t)(r0 + m) * F_DIM + k + hi * 2));
        v2f b;
        b.x = edgeW[(size_t)(k + hi * 2)     * F_DIM + cg];
        b.y = edgeW[(size_t)(k + hi * 2 + 1) * F_DIM + cg];
        acc = __builtin_amdgcn_wmma_f32_16x16x4_f32(false, a, false, b, (short)0, acc, false, false);
    }
#else
    // VALU fallback (keeps compile alive if the f32 WMMA builtin differs)
#pragma unroll
    for (int v = 0; v < 8; ++v) {
        const int row = r0 + (hi << 3) + v;
        float s = 0.f;
        for (int j = 0; j < 128; ++j)
            s += posAgg[(size_t)row * F_DIM + j] * nodeW[(size_t)j * F_DIM + cg];
        for (int j = 0; j < 128; ++j)
            s += negAgg[(size_t)row * F_DIM + j] * nodeW[(size_t)(j + 128) * F_DIM + cg];
        for (int j = 0; j < 128; ++j)
            s += edgeAgg[(size_t)row * F_DIM + j] * edgeW[(size_t)j * F_DIM + cg];
        acc[v] = s;
    }
#endif

    const float nb = nodeB[cg];
    const float eb = edgeB[cg];
#pragma unroll
    for (int v = 0; v < 8; ++v) {
        const int row = r0 + (hi << 3) + v;
        const float dE = edge_adj[(size_t)row * N_DIM + row];
        const float dN = node_adj[(size_t)row * N_DIM + row];
        out[(size_t)row * F_DIM + cg] =
            acc[v] + nb * dE + eb * dN + feats[(size_t)row * F_DIM + cg];
    }
}

// ---------------------------------------------------------------------------
extern "C" void kernel_launch(void* const* d_in, const int* in_sizes, int n_in,
                              void* d_out, int out_size, void* d_ws, size_t ws_size,
                              hipStream_t stream) {
    const float* feats    = (const float*)d_in[0];
    const float* node_adj = (const float*)d_in[1];
    const float* edge_adj = (const float*)d_in[2];
    const float* node_w   = (const float*)d_in[3];
    const float* node_b   = (const float*)d_in[4];
    const float* edge_w   = (const float*)d_in[5];
    const float* edge_b   = (const float*)d_in[6];
    float*       out      = (float*)d_out;

    char* ws = (char*)d_ws;
    unsigned short* ftHi = (unsigned short*)(ws);                 //  2 MB
    unsigned short* ftLo = (unsigned short*)(ws + (2u << 20));    //  2 MB
    float* posAgg  = (float*)(ws + (4u  << 20));                  //  4 MB
    float* negAgg  = (float*)(ws + (8u  << 20));                  //  4 MB
    float* edgeAgg = (float*)(ws + (12u << 20));                  //  4 MB

    prep_featsT<<<(N_DIM * F_DIM) / 256, 256, 0, stream>>>(feats, ftHi, ftLo);

    agg_kernel<<<N_DIM / 32, 256, 0, stream>>>(node_adj, edge_adj, ftHi, ftLo,
                                               posAgg, negAgg, edgeAgg);

    out_kernel<<<N_DIM / 16, 256, 0, stream>>>(posAgg, negAgg, edgeAgg,
                                               node_w, edge_w, node_b, edge_b,
                                               node_adj, edge_adj, feats, out);

    (void)in_sizes; (void)n_in; (void)out_size; (void)ws_size;
}
